// MaskedAttention_67413806678107
// MI455X (gfx1250) — matrix operand order repaired and checked by reading, not compiled, not run on hardware
//
#include <hip/hip_runtime.h>

// ---------------------------------------------------------------------------
// Causal masked attention, MI455X (gfx1250, wave32, WMMA bf16).
//   Phase 1: QKV projections -> bf16 workspace (V stored transposed [D][S]).
//   Phase 2: flash attention, 16-query blocks, 128-key blocks, online softmax.
// Workspace: 3 * 8192*1024 bf16 = 48 MB (fits entirely in the 192 MB L2).
// ---------------------------------------------------------------------------

#define BB 4
#define SS 2048
#define DD 1024
#define MM (BB * SS)

typedef __attribute__((ext_vector_type(16))) __bf16       v16bf;
typedef __attribute__((ext_vector_type(8)))  float        v8f;
typedef __attribute__((ext_vector_type(4)))  unsigned int u32x4;
typedef __attribute__((ext_vector_type(4)))  float        f32x4;

union BF16Frag { v16bf bf; u32x4 q[2]; };
union BF16Pack { u32x4 q; unsigned short u[8]; };

static __device__ __forceinline__ unsigned pack2(float lo, float hi) {
  unsigned short a = __builtin_bit_cast(unsigned short, (__bf16)lo);
  unsigned short b = __builtin_bit_cast(unsigned short, (__bf16)hi);
  return (unsigned)a | ((unsigned)b << 16);
}

// ---------------------------------------------------------------------------
// Phase 1: Y = X @ W for Wq/Wk/Wv.  Tile 64(M) x 128(N), K-step 32.
// 8 waves as 2x4; each wave owns a 32x32 block = 2x2 WMMA tiles.
// LDS row stride 40 bf16 (80 B): 16-lane b128 A reads touch all 64 banks once.
// ---------------------------------------------------------------------------
#define AST 40
#define WST 40

__global__ __launch_bounds__(256) void qkv_gemm(
    const float* __restrict__ x,  const float* __restrict__ Wq,
    const float* __restrict__ Wk, const float* __restrict__ Wv,
    __bf16* __restrict__ Qo, __bf16* __restrict__ Ko, __bf16* __restrict__ Vt)
{
  __shared__ __bf16 As[64 * AST];    // X tile  [m][k]
  __shared__ __bf16 Ws[128 * WST];   // W tile transposed: [n][k]

  const int which = blockIdx.z;
  const float* __restrict__ W = (which == 0) ? Wq : (which == 1) ? Wk : Wv;
  const int m0 = blockIdx.x * 64;
  const int n0 = blockIdx.y * 128;
  const int tid  = threadIdx.x;
  const int lane = tid & 31;
  const int wave = tid >> 5;
  const int wm = wave >> 2;          // 0..1
  const int wn = wave & 3;           // 0..3
  const int lm   = lane & 15;
  const int half = lane >> 4;

  // staging coordinates (loop-invariant)
  const int xrow = tid >> 2, xc0 = (tid & 3) * 8;            // X tile: 8 elems
  const int wkk  = (tid & 15) * 2, wnc = (tid >> 4) * 8;     // W tile: 2k x 8n

  v8f acc[2][2] = {};

  for (int k0 = 0; k0 < DD; k0 += 32) {
    // ---- stage X tile 64x32: fp32 -> bf16, one ds_store_b128 / thread ----
    {
      const float* src = x + (size_t)(m0 + xrow) * DD + k0 + xc0;
      f32x4 a = *(const f32x4*)(src);
      f32x4 b = *(const f32x4*)(src + 4);
      if (k0 + 32 < DD) __builtin_prefetch(src + 32, 0, 1);
      BF16Pack pk;
#pragma unroll
      for (int j = 0; j < 4; ++j) {
        pk.u[j]     = __builtin_bit_cast(unsigned short, (__bf16)a[j]);
        pk.u[4 + j] = __builtin_bit_cast(unsigned short, (__bf16)b[j]);
      }
      *(u32x4*)&As[xrow * AST + xc0] = pk.q;
    }
    // ---- stage W tile 32x128 transposed: pack k-pairs, ds_store_b32 ----
    {
      const float* s0 = W + (size_t)(k0 + wkk) * DD + n0 + wnc;
      const float* s1 = s0 + DD;
      f32x4 r0a = *(const f32x4*)(s0), r0b = *(const f32x4*)(s0 + 4);
      f32x4 r1a = *(const f32x4*)(s1), r1b = *(const f32x4*)(s1 + 4);
      if (k0 + 32 < DD) __builtin_prefetch(s0 + 32 * DD, 0, 1);
#pragma unroll
      for (int j = 0; j < 4; ++j) {
        *(unsigned*)&Ws[(wnc + j) * WST + wkk]     = pack2(r0a[j], r1a[j]);
        *(unsigned*)&Ws[(wnc + 4 + j) * WST + wkk] = pack2(r0b[j], r1b[j]);
      }
    }
    __syncthreads();

    BF16Frag a[2], bfr[2];
#pragma unroll
    for (int mt = 0; mt < 2; ++mt) {
      const __bf16* p = &As[(32 * wm + 16 * mt + lm) * AST + half * 16];
      a[mt].q[0] = *(const u32x4*)(p);
      a[mt].q[1] = *(const u32x4*)(p + 8);
    }
#pragma unroll
    for (int nt = 0; nt < 2; ++nt) {
      const __bf16* p = &Ws[(32 * wn + 16 * nt + lm) * WST + half * 16];
      bfr[nt].q[0] = *(const u32x4*)(p);
      bfr[nt].q[1] = *(const u32x4*)(p + 8);
    }
#pragma unroll
    for (int mt = 0; mt < 2; ++mt)
#pragma unroll
      for (int nt = 0; nt < 2; ++nt)
        acc[mt][nt] = __builtin_amdgcn_wmma_f32_16x16x32_bf16(
            false, a[mt].bf, false, bfr[nt].bf, (short)0, acc[mt][nt], false, false);
    __syncthreads();
  }

  // ---- epilogue: uniform `which` hoisted out of the store loops ----
  // C layout: row = r + 8*half, col = lane&15
  if (which < 2) {
    __bf16* __restrict__ O = (which == 0) ? Qo : Ko;
#pragma unroll
    for (int mt = 0; mt < 2; ++mt)
#pragma unroll
      for (int nt = 0; nt < 2; ++nt) {
        const size_t base =
            (size_t)(m0 + 32 * wm + 16 * mt + 8 * half) * DD +
            (n0 + 32 * wn + 16 * nt + lm);
#pragma unroll
        for (int r = 0; r < 8; ++r)
          O[base + (size_t)r * DD] = (__bf16)acc[mt][nt][r];
      }
  } else {
    // Vt[b][d][s]: each lane's 8 rows are 8 consecutive s at fixed d
    const int bb = m0 >> 11;                    // tile never crosses a batch
    const int sb = (m0 & (SS - 1)) + 32 * wm + 8 * half;
#pragma unroll
    for (int mt = 0; mt < 2; ++mt)
#pragma unroll
      for (int nt = 0; nt < 2; ++nt) {
        const int col = n0 + 32 * wn + 16 * nt + lm;
        BF16Pack pk;
#pragma unroll
        for (int r = 0; r < 8; ++r)
          pk.u[r] = __builtin_bit_cast(unsigned short, (__bf16)acc[mt][nt][r]);
        *(u32x4*)&Vt[((size_t)bb * DD + col) * SS + sb + 16 * mt] = pk.q;
      }
  }
}

// ---------------------------------------------------------------------------
// Phase 2: flash attention. One workgroup = 16 queries of one batch.
// Key blocks of 128; wave w computes score tile for keys [k0+16w, k0+16w+16)
// and owns output D-chunk [128w, 128w+128) (acc = 8 x v8f).
// ---------------------------------------------------------------------------
#define QST 1032   // bf16 stride: 2064 B, 16B-aligned rows
#define SST 132    // f32 stride
#define PST 136    // bf16 stride (272 B, 16B-aligned)

__global__ __launch_bounds__(256) void attn(
    const __bf16* __restrict__ Q, const __bf16* __restrict__ K,
    const __bf16* __restrict__ Vt, float* __restrict__ out)
{
  __shared__ __bf16 Qs[16 * QST];
  __shared__ float  Ss[16 * SST];
  __shared__ __bf16 Ps[16 * PST];
  __shared__ float  mrow[16], lrow[16], frow[16];

  const int b  = blockIdx.y;
  const int q0 = blockIdx.x * 16;
  const int tid  = threadIdx.x;
  const int lane = tid & 31;
  const int wave = tid >> 5;
  const int lm   = lane & 15;
  const int half = lane >> 4;

  const __bf16* __restrict__ Qb = Q  + ((size_t)b * SS + q0) * DD;
  const __bf16* __restrict__ Kb = K  + (size_t)b * SS * DD;
  const __bf16* __restrict__ Vb = Vt + (size_t)b * DD * SS;

  // stage the 16x1024 Q block once
  for (int i = tid; i < 16 * DD / 8; i += 256) {
    int row = i >> 7;
    int c   = (i & 127) * 8;
    *(u32x4*)&Qs[row * QST + c] = *(const u32x4*)&Qb[(size_t)row * DD + c];
  }
  if (tid < 16) { mrow[tid] = -1e30f; lrow[tid] = 0.f; }
  __syncthreads();

  v8f acc[8] = {};
  const float scale = 0.03125f;               // 1/sqrt(1024)
  const int nkb = (q0 >> 7) + 1;              // causal: skip blocks past diag

  for (int kb = 0; kb < nkb; ++kb) {
    const int k0 = kb * 128;

    // ---- scores: S_tile = Q(16x1024) . K^T, K-dim = feature e ----
    v8f sc = {};
    const __bf16* Kp = Kb + (size_t)(k0 + 16 * wave + lm) * DD + half * 16;
    if (kb + 1 < nkb) __builtin_prefetch(Kp + (size_t)128 * DD, 0, 1);
    for (int e = 0; e < DD; e += 32) {
      BF16Frag a, bbf;
      const __bf16* qp = &Qs[lm * QST + e + half * 16];
      a.q[0]   = *(const u32x4*)(qp);
      a.q[1]   = *(const u32x4*)(qp + 8);
      bbf.q[0] = *(const u32x4*)(Kp + e);
      bbf.q[1] = *(const u32x4*)(Kp + e + 8);
      sc = __builtin_amdgcn_wmma_f32_16x16x32_bf16(false, a.bf, false, bbf.bf,
                                                   (short)0, sc, false, false);
    }
#pragma unroll
    for (int r = 0; r < 8; ++r)
      Ss[(r + 8 * half) * SST + 16 * wave + lm] = sc[r];
    __syncthreads();

    // ---- online softmax: each half-wave owns one row (8 cols / lane) ----
    {
      int row = 2 * wave + half;
      int c0  = lm * 8;
      float vals[8], vmax = -1e30f;
#pragma unroll
      for (int j = 0; j < 8; ++j) {
        float s = Ss[row * SST + c0 + j] * scale;
        if (k0 + c0 + j > q0 + row) s = -1e30f;       // causal mask
        vals[j] = s;
        vmax = fmaxf(vmax, s);
      }
#pragma unroll
      for (int off = 1; off < 16; off <<= 1)
        vmax = fmaxf(vmax, __shfl_xor(vmax, off, 32));
      float mold = mrow[row];
      float mnew = fmaxf(mold, vmax);
      float fac  = __expf(mold - mnew);
      float psum = 0.f;
      BF16Pack pk;
#pragma unroll
      for (int j = 0; j < 8; ++j) {
        float p = __expf(vals[j] - mnew);
        psum += p;
        pk.u[j] = __builtin_bit_cast(unsigned short, (__bf16)p);
      }
      *(u32x4*)&Ps[row * PST + c0] = pk.q;            // one ds_store_b128
#pragma unroll
      for (int off = 1; off < 16; off <<= 1)
        psum += __shfl_xor(psum, off, 32);
      if (lm == 0) {
        mrow[row] = mnew;
        lrow[row] = lrow[row] * fac + psum;
        frow[row] = fac;
      }
    }
    __syncthreads();

    // ---- rescale acc, then acc += P(16x128) . V_chunk(128 x 128w..) ----
    {
      v8f facv;
#pragma unroll
      for (int r = 0; r < 8; ++r) facv[r] = frow[r + 8 * half];
#pragma unroll
      for (int nt = 0; nt < 8; ++nt) acc[nt] *= facv;

      for (int ks = 0; ks < 128; ks += 32) {
        BF16Frag a;
        const __bf16* pp = &Ps[lm * PST + ks + half * 16];
        a.q[0] = *(const u32x4*)(pp);
        a.q[1] = *(const u32x4*)(pp + 8);
#pragma unroll
        for (int nt = 0; nt < 8; ++nt) {
          BF16Frag bbf;
          const __bf16* vp =
              Vb + (size_t)(128 * wave + 16 * nt + lm) * SS + k0 + ks + half * 16;
          bbf.q[0] = *(const u32x4*)(vp);
          bbf.q[1] = *(const u32x4*)(vp + 8);
          acc[nt] = __builtin_amdgcn_wmma_f32_16x16x32_bf16(
              false, a.bf, false, bbf.bf, (short)0, acc[nt], false, false);
        }
      }
    }
    __syncthreads();
  }

  // ---- epilogue: out = acc / l ----
  v8f linv;
#pragma unroll
  for (int r = 0; r < 8; ++r) linv[r] = 1.0f / lrow[r + 8 * half];
#pragma unroll
  for (int nt = 0; nt < 8; ++nt) {
    v8f o = acc[nt] * linv;
#pragma unroll
    for (int r = 0; r < 8; ++r) {
      int row = q0 + r + 8 * half;
      int col = 128 * wave + 16 * nt + lm;
      out[((size_t)b * SS + row) * DD + col] = o[r];
    }
  }
}

// ---------------------------------------------------------------------------
extern "C" void kernel_launch(void* const* d_in, const int* in_sizes, int n_in,
                              void* d_out, int out_size, void* d_ws, size_t ws_size,
                              hipStream_t stream) {
  const float* x  = (const float*)d_in[0];
  const float* Wq = (const float*)d_in[1];
  const float* Wk = (const float*)d_in[2];
  const float* Wv = (const float*)d_in[3];
  float* out = (float*)d_out;

  __bf16* Qw = (__bf16*)d_ws;                 // [M][D] bf16
  __bf16* Kw = Qw + (size_t)MM * DD;          // [M][D] bf16
  __bf16* Vw = Kw + (size_t)MM * DD;          // [B][D][S] bf16 (transposed)

  dim3 g1(MM / 64, DD / 128, 3);
  qkv_gemm<<<g1, 256, 0, stream>>>(x, Wq, Wk, Wv, Qw, Kw, Vw);

  dim3 g2(SS / 16, BB, 1);
  attn<<<g2, 256, 0, stream>>>(Qw, Kw, Vw, out);
}
